// DynamicCRF_6777458393848
// MI455X (gfx1250) — compile-verified, matrix operand order checked by reading, not run on hardware
//
#include <hip/hip_runtime.h>

#define BEAM  64
#define RANK  32
#define VOCAB 21128
#define BATCH 16
#define SEQ   256

typedef __attribute__((ext_vector_type(16))) _Float16 v16h;
typedef __attribute__((ext_vector_type(8)))  float    v8f;

// ---------------------------------------------------------------------------
// Kernel A: exact top-64 selection per (b,s) via LDS-resident radix select.
// Deterministic: histograms are order-independent counts; compaction offsets
// come from a serial exclusive scan, so output positions are reproducible.
// ---------------------------------------------------------------------------
__global__ void topk64_kernel(const float* __restrict__ emissions,
                              const int*   __restrict__ targets,
                              int*         __restrict__ beam_targets,
                              float*       __restrict__ beam_em)
{
    extern __shared__ unsigned int smem[];
    unsigned int* s_keys = smem;               // VOCAB
    unsigned int* s_hist = s_keys + VOCAB;     // 256
    unsigned int* s_cgt  = s_hist + 256;       // 256
    unsigned int* s_ceq  = s_cgt  + 256;       // 256
    unsigned int* s_out  = s_ceq  + 256;       // 64
    unsigned int* s_misc = s_out  + 64;        // [0]=prefix [1]=rank

    const int s = blockIdx.x;
    const int b = blockIdx.y;
    const int t = threadIdx.x;                 // 256 threads

    const float* row = emissions + ((size_t)b * SEQ + s) * (size_t)VOCAB;
    const int tgt = targets[b * SEQ + s];

    // Load row, map to order-preserving unsigned keys.
    for (int v = t; v < VOCAB; v += 256) {
        unsigned int u = __float_as_uint(row[v]);
        s_keys[v] = (u & 0x80000000u) ? ~u : (u | 0x80000000u);
    }
    __syncthreads();
    if (t == 0) {
        s_keys[tgt] = 0xFFFFFFFFu;             // force gold target into beam
        s_misc[0] = 0u;                        // prefix
        s_misc[1] = BEAM;                      // rank of threshold from top
    }
    __syncthreads();

    // 4 byte-passes: find the 64th-largest key T.
    for (int pass = 0; pass < 4; ++pass) {
        const int shift = 24 - 8 * pass;
        const unsigned int himask = (pass == 0) ? 0u : (0xFFFFFFFFu << (shift + 8));
        const unsigned int prefix = s_misc[0];
        if (t < 256) s_hist[t] = 0u;
        __syncthreads();
        for (int v = t; v < VOCAB; v += 256) {
            unsigned int k = s_keys[v];
            if ((k & himask) == prefix)
                atomicAdd(&s_hist[(k >> shift) & 255u], 1u);
        }
        __syncthreads();
        if (t == 0) {
            unsigned int r = s_misc[1];
            unsigned int p = s_misc[0];
            for (int bin = 255; bin >= 0; --bin) {
                unsigned int c = s_hist[bin];
                if (r <= c) { p |= ((unsigned int)bin) << shift; break; }
                r -= c;
            }
            s_misc[0] = p; s_misc[1] = r;
        }
        __syncthreads();
    }

    const unsigned int T    = s_misc[0];
    const unsigned int rfin = s_misc[1];       // how many ==T to take
    const unsigned int G    = BEAM - rfin;     // count strictly greater than T

    // Deterministic compaction of {keys > T} then {keys == T} (capped).
    unsigned int cg = 0, ce = 0;
    for (int v = t; v < VOCAB; v += 256) {
        unsigned int k = s_keys[v];
        cg += (k > T); ce += (k == T);
    }
    s_cgt[t] = cg; s_ceq[t] = ce;
    __syncthreads();
    if (t == 0) {
        unsigned int ag = 0, ae = G;
        for (int i = 0; i < 256; ++i) {
            unsigned int g = s_cgt[i]; s_cgt[i] = ag; ag += g;
            unsigned int e = s_ceq[i]; s_ceq[i] = ae; ae += e;
        }
    }
    __syncthreads();
    unsigned int pg = s_cgt[t], pe = s_ceq[t];
    for (int v = t; v < VOCAB; v += 256) {
        unsigned int k = s_keys[v];
        if (k > T)       { s_out[pg++] = (unsigned int)v; }
        else if (k == T) { if (pe < BEAM) s_out[pe] = (unsigned int)v; ++pe; }
    }
    __syncthreads();

    if (t < BEAM) {
        int idx = (int)s_out[t];
        size_t o = (((size_t)b * SEQ) + s) * BEAM + t;
        beam_targets[o] = idx;
        beam_em[o] = row[idx];                 // real emission (not the inf mark)
    }
}

// ---------------------------------------------------------------------------
// Kernel B: one workgroup (8 wave32) per batch. Numerator + fused
// (WMMA beam-transition GEMM -> logsumexp) forward scan.
// Per step: T = E1[prev_beam] (64x32) x E2[cur_beam]^T (32x64) via 16
// v_wmma_f32_16x16x32_f16, then nxt[j] = LSE_i(score_i + T_ij) + emit_j.
// ---------------------------------------------------------------------------
__global__ void crf_scan_kernel(const float* __restrict__ emissions,
                                const int*   __restrict__ targets,
                                const unsigned char* __restrict__ mask,
                                const float* __restrict__ E1,
                                const float* __restrict__ E2,
                                const int*   __restrict__ beam_targets,
                                const float* __restrict__ beam_em,
                                float*       __restrict__ llh)
{
    __shared__ _Float16 s_A[BEAM * RANK];     // E1 rows of prev beam (f16)
    __shared__ _Float16 s_B[BEAM * RANK];     // E2 rows of cur beam (f16)
    __shared__ float    s_T[BEAM * 68];       // 64x64 padded to stride 68
    __shared__ float    s_score[BEAM];
    __shared__ float    s_red[4 * BEAM];
    __shared__ float    s_m[BEAM];
    __shared__ float    s_part[256];
    __shared__ float    s_num;

    const int b    = blockIdx.x;
    const int t    = threadIdx.x;             // 256 threads = SEQ
    const int lane = t & 31;
    const int wave = t >> 5;

    // ---- numerator: em[s]*mask[s] + trans[s-1]*mask[s] ----
    {
        const int sidx = t;
        const int tg = targets[b * SEQ + sidx];
        const float m = mask[b * SEQ + sidx] ? 1.0f : 0.0f;
        float part = emissions[(((size_t)b * SEQ) + sidx) * (size_t)VOCAB + tg] * m;
        if (sidx >= 1) {
            const int tp = targets[b * SEQ + sidx - 1];
            float dot = 0.0f;
            #pragma unroll
            for (int k = 0; k < RANK; ++k)
                dot += E1[tp * RANK + k] * E2[tg * RANK + k];
            part += dot * m;
        }
        s_part[t] = part;
    }
    __syncthreads();
    if (t == 0) {
        float a = 0.0f;
        for (int i = 0; i < 256; ++i) a += s_part[i];   // fixed order
        s_num = a;
    }

    // ---- score0 = beam_em[:,0] ----
    if (t < BEAM) s_score[t] = beam_em[((size_t)b * SEQ) * BEAM + t];
    __syncthreads();

    for (int s = 1; s < SEQ; ++s) {
        const int* Pb = beam_targets + (((size_t)b * SEQ) + (s - 1)) * BEAM;
        const int* Cb = beam_targets + (((size_t)b * SEQ) + s) * BEAM;

        // Stage gathered rows as f16 (coalesced across k).
        for (int e = t; e < BEAM * RANK; e += 256) {
            const int rrow = e >> 5, k = e & 31;
            s_A[e] = (_Float16)E1[Pb[rrow] * RANK + k];
            s_B[e] = (_Float16)E2[Cb[rrow] * RANK + k];
        }
        __syncthreads();

        // 16 output tiles, 2 per wave.  K=32 == one WMMA per tile.
        #pragma unroll
        for (int q = 0; q < 2; ++q) {
            const int tid = wave * 2 + q;
            const int ti = tid >> 2, tj = tid & 3;
            const int kb0 = (lane < 16) ? 0 : 8;   // ISA 16-bit A/B layout
            const int kb1 = kb0 + 16;
            const _Float16* Ar = &s_A[(ti * 16 + (lane & 15)) * RANK];
            const _Float16* Br = &s_B[(tj * 16 + (lane & 15)) * RANK];
            v16h a, bb;
            #pragma unroll
            for (int p = 0; p < 8; ++p) {
                a[p]      = Ar[kb0 + p];
                a[8 + p]  = Ar[kb1 + p];
                bb[p]     = Br[kb0 + p];
                bb[8 + p] = Br[kb1 + p];
            }
            v8f c = {0.f, 0.f, 0.f, 0.f, 0.f, 0.f, 0.f, 0.f};
            v8f d = __builtin_amdgcn_wmma_f32_16x16x32_f16(
                        false, a, false, bb, (short)0, c, false, false);
            const int i0 = ti * 16 + ((lane < 16) ? 0 : 8);
            const int j  = tj * 16 + (lane & 15);
            #pragma unroll
            for (int r = 0; r < 8; ++r)
                s_T[(i0 + r) * 68 + j] = d[r];      // stride 68: bank-clean
        }
        __syncthreads();

        // Max-subtracted logsumexp down columns. 4 partials x 64 columns.
        {
            const int j = t & 63, p = t >> 6;
            float mx = -3.402823466e38f;
            #pragma unroll
            for (int i = p * 16; i < p * 16 + 16; ++i)
                mx = fmaxf(mx, s_score[i] + s_T[i * 68 + j]);
            s_red[p * BEAM + j] = mx;
        }
        __syncthreads();
        if (t < BEAM) {
            float mx = s_red[t];
            #pragma unroll
            for (int p = 1; p < 4; ++p) mx = fmaxf(mx, s_red[p * BEAM + t]);
            s_m[t] = mx;
        }
        __syncthreads();
        {
            const int j = t & 63, p = t >> 6;
            const float mj = s_m[j];
            float sum = 0.0f;
            #pragma unroll
            for (int i = p * 16; i < p * 16 + 16; ++i)
                sum += __expf(s_score[i] + s_T[i * 68 + j] - mj);
            s_red[p * BEAM + j] = sum;
        }
        __syncthreads();
        if (t < BEAM) {
            float tot = 0.0f;
            #pragma unroll
            for (int p = 0; p < 4; ++p) tot += s_red[p * BEAM + t];  // fixed order
            const float emit = beam_em[(((size_t)b * SEQ) + s) * BEAM + t];
            const float nxt = s_m[t] + __logf(tot) + emit;
            if (mask[b * SEQ + s]) s_score[t] = nxt;
        }
        __syncthreads();
    }

    if (t == 0) {
        float mx = s_score[0];
        for (int j = 1; j < BEAM; ++j) mx = fmaxf(mx, s_score[j]);
        float sum = 0.0f;
        for (int j = 0; j < BEAM; ++j) sum += __expf(s_score[j] - mx);
        llh[b] = s_num - (mx + __logf(sum));
    }
}

// ---------------------------------------------------------------------------
// Kernel C: deterministic 16-way sum -> scalar output.
// ---------------------------------------------------------------------------
__global__ void finalize_kernel(const float* __restrict__ llh,
                                float* __restrict__ out)
{
    if (threadIdx.x == 0) {
        float a = 0.0f;
        for (int b = 0; b < BATCH; ++b) a += llh[b];
        out[0] = a;
    }
}

extern "C" void kernel_launch(void* const* d_in, const int* in_sizes, int n_in,
                              void* d_out, int out_size, void* d_ws, size_t ws_size,
                              hipStream_t stream)
{
    (void)in_sizes; (void)n_in; (void)out_size; (void)ws_size;
    const float*         emissions = (const float*)d_in[0];
    const int*           targets   = (const int*)d_in[1];
    const unsigned char* mask      = (const unsigned char*)d_in[2]; // jax bool = 1B
    const float*         E1        = (const float*)d_in[3];
    const float*         E2        = (const float*)d_in[4];

    char* ws = (char*)d_ws;
    const size_t beamElems = (size_t)BATCH * SEQ * BEAM;
    int*   beam_targets = (int*)ws;                              // 1 MB
    float* beam_em      = (float*)(ws + beamElems * 4);          // 1 MB
    float* llh          = (float*)(ws + 2 * beamElems * 4);      // 64 B

    const size_t ldsA = (VOCAB + 3 * 256 + 64 + 16) * sizeof(unsigned int);
    topk64_kernel<<<dim3(SEQ, BATCH), dim3(256), ldsA, stream>>>(
        emissions, targets, beam_targets, beam_em);

    crf_scan_kernel<<<dim3(BATCH), dim3(256), 0, stream>>>(
        emissions, targets, mask, E1, E2, beam_targets, beam_em, llh);

    finalize_kernel<<<dim3(1), dim3(32), 0, stream>>>(llh, (float*)d_out);
}
